// TernaryMLP_33200097198453
// MI455X (gfx1250) — compile-verified, matrix operand order checked by reading, not run on hardware
//
#include <hip/hip_runtime.h>

typedef __attribute__((ext_vector_type(16))) _Float16 v16h;
typedef __attribute__((ext_vector_type(8)))  float    v8f;
typedef __attribute__((ext_vector_type(4)))  _Float16 v4h;
typedef __attribute__((ext_vector_type(4)))  int      v4i;

#define LDSS 72   // LDS row stride in halfs (64 + 8 pad) -> 144B = 36 banks, conflict-free b128 reads
#define KT   64   // K-tile per LDS stage (two WMMA k-steps of 32)

#define HAS_ASYNC_LDS (__has_builtin(__builtin_amdgcn_global_load_async_to_lds_b128))

// ---- CDNA5 async copy: global -> LDS without VGPR round-trip (ASYNCcnt-tracked) ----
__device__ __forceinline__ void async_cp16(const _Float16* g, _Float16* l) {
#if HAS_ASYNC_LDS
    __builtin_amdgcn_global_load_async_to_lds_b128(
        (__attribute__((address_space(1))) v4i*)g,
        (__attribute__((address_space(3))) v4i*)l,
        0, 0);
#else
    *(float4*)l = *(const float4*)g;   // fallback relay through VGPRs
#endif
}

template <int N>
__device__ __forceinline__ void wait_async() {
#if __has_builtin(__builtin_amdgcn_s_wait_asynccnt)
    __builtin_amdgcn_s_wait_asynccnt(N);
#else
    if constexpr (N == 0) asm volatile("s_wait_asynccnt 0x0" ::: "memory");
    else                  asm volatile("s_wait_asynccnt 0x8" ::: "memory");
#endif
}

// ---------------- abs-sum reduction (stage 1: per-block partials) ----------------
__global__ __launch_bounds__(256)
void absum_partial(const float* __restrict__ w, long n, float* __restrict__ partial) {
    __shared__ float red[256];
    float s = 0.0f;
    for (long i = (long)blockIdx.x * 256 + threadIdx.x; i < n; i += (long)gridDim.x * 256)
        s += fabsf(w[i]);
    red[threadIdx.x] = s;
    __syncthreads();
    for (int off = 128; off > 0; off >>= 1) {
        if ((int)threadIdx.x < off) red[threadIdx.x] += red[threadIdx.x + off];
        __syncthreads();
    }
    if (threadIdx.x == 0) partial[blockIdx.x] = red[0];
}

// ---------------- stage 2: deterministic finalize -> gamma = mean(|W|) ----------------
__global__ __launch_bounds__(256)
void finalize_gamma(const float* __restrict__ partials, float* __restrict__ gammas,
                    float inv0, float inv1, float inv2, float inv3) {
    __shared__ float red[256];
    const int mat = blockIdx.x;
    const float* p = partials + (size_t)mat * 1024;
    float s = p[threadIdx.x] + p[threadIdx.x + 256] + p[threadIdx.x + 512] + p[threadIdx.x + 768];
    red[threadIdx.x] = s;
    __syncthreads();
    for (int off = 128; off > 0; off >>= 1) {
        if ((int)threadIdx.x < off) red[threadIdx.x] += red[threadIdx.x + off];
        __syncthreads();
    }
    if (threadIdx.x == 0) {
        float inv = (mat == 0) ? inv0 : (mat == 1) ? inv1 : (mat == 2) ? inv2 : inv3;
        gammas[mat] = red[0] * inv;
    }
}

// ---------------- ternary quantize: T = clip(rint(w/(gamma+eps))), stored f16 ----------------
__global__ __launch_bounds__(256)
void quantize_tern(const float* __restrict__ w, _Float16* __restrict__ t,
                   long n_real, long n_alloc, const float* __restrict__ gptr) {
    const float g = *gptr + 1e-5f;
    const long stride = (long)gridDim.x * 256;
    for (long i = (long)blockIdx.x * 256 + threadIdx.x; i < n_alloc; i += stride) {
        float q = 0.0f;
        if (i < n_real) {
            q = rintf(w[i] / g);
            q = fmaxf(-1.0f, fminf(1.0f, q));
        }
        t[i] = (_Float16)q;
    }
}

// ---------------- f32 -> f16 convert (vectorized x4) ----------------
__global__ __launch_bounds__(256)
void cvt_f16(const float* __restrict__ x, _Float16* __restrict__ y, long n4) {
    long i = (long)blockIdx.x * 256 + threadIdx.x;
    if (i < n4) {
        float4 v = ((const float4*)x)[i];
        v4h h;
        h[0] = (_Float16)v.x; h[1] = (_Float16)v.y;
        h[2] = (_Float16)v.z; h[3] = (_Float16)v.w;
        ((v4h*)y)[i] = h;
    }
}

// ---------------- WMMA GEMM: out = act( gamma * (A @ T^T) + bias ) ----------------
// A: [M,K] f16 row-major; T: [Npad,K] f16 row-major (row n = B-column n).
// Block tile 128x128, 8 waves (wave32), per-wave 64x32 = 4x2 fragments of 16x16.
// Triple-buffered LDS stages via GLOBAL_LOAD_ASYNC_TO_LDS_B128; single barrier/iter.
template <bool RELU, typename OutT>
__global__ __launch_bounds__(256)
void gemm_tern(const _Float16* __restrict__ A, const _Float16* __restrict__ W,
               const float* __restrict__ bias, const float* __restrict__ gptr,
               OutT* __restrict__ out, int M, int Nreal, int K, int ldc) {
    __shared__ _Float16 sA[3][128 * LDSS];
    __shared__ _Float16 sB[3][128 * LDSS];

    const int tid  = threadIdx.x;
    const int lane = tid & 31;
    const int wave = tid >> 5;
    const int wm   = (wave & 1) * 64;   // wave offset in M within tile
    const int wn   = (wave >> 1) * 32;  // wave offset in N within tile
    const int bm   = blockIdx.y * 128;
    const int bn   = blockIdx.x * 128;
    const int r16  = lane & 15;
    const int hi   = lane >> 4;         // lane-half select (K/M interleave)

    auto issue_stage = [&](int buf, int kk) {
#pragma unroll
        for (int i = 0; i < 4; ++i) {
            const int chunk = tid + i * 256;
            const int row   = chunk >> 3;
            const int col   = (chunk & 7) * 8;
            async_cp16(A + (size_t)(bm + row) * K + kk + col, &sA[buf][row * LDSS + col]);
            async_cp16(W + (size_t)(bn + row) * K + kk + col, &sB[buf][row * LDSS + col]);
        }
    };

    v8f acc[4][2];
#pragma unroll
    for (int mf = 0; mf < 4; ++mf)
#pragma unroll
        for (int nf = 0; nf < 2; ++nf)
#pragma unroll
            for (int r = 0; r < 8; ++r) acc[mf][nf][r] = 0.0f;

    const int nstages = K / KT;   // K is a multiple of KT; nstages >= 2 here

    // prologue: two stages in flight
    issue_stage(0, 0);
    issue_stage(1, KT);

    int cur = 0;
    for (int it = 0; it < nstages; ++it) {
        if (it + 1 < nstages) wait_async<8>();   // stage `it` done; `it+1` still in flight
        else                  wait_async<0>();
        __syncthreads();                         // stage data visible; stage it-1 reads done

        if (it + 2 < nstages) {
            int nxt = cur + 2; if (nxt >= 3) nxt -= 3;
            issue_stage(nxt, (it + 2) * KT);     // overwrites stage it-1's buffer: safe
        }

        const _Float16* tA = sA[cur];
        const _Float16* tB = sB[cur];
#pragma unroll
        for (int ks = 0; ks < KT; ks += 32) {
            const int cbase = ks + hi * 8;       // ISA 16-bit fragment K-interleave
            v16h a[4], b[2];
#pragma unroll
            for (int mf = 0; mf < 4; ++mf) {
                const _Float16* q = tA + (size_t)(wm + mf * 16 + r16) * LDSS + cbase;
                ((float4*)&a[mf])[0] = *(const float4*)(q);
                ((float4*)&a[mf])[1] = *(const float4*)(q + 16);
            }
#pragma unroll
            for (int nf = 0; nf < 2; ++nf) {
                const _Float16* q = tB + (size_t)(wn + nf * 16 + r16) * LDSS + cbase;
                ((float4*)&b[nf])[0] = *(const float4*)(q);
                ((float4*)&b[nf])[1] = *(const float4*)(q + 16);
            }
#pragma unroll
            for (int mf = 0; mf < 4; ++mf)
#pragma unroll
                for (int nf = 0; nf < 2; ++nf)
                    acc[mf][nf] = __builtin_amdgcn_wmma_f32_16x16x32_f16(
                        false, a[mf], false, b[nf], (short)0, acc[mf][nf], false, false);
        }

        cur = (cur == 2) ? 0 : cur + 1;
    }

    const float gamma = *gptr;
#pragma unroll
    for (int mf = 0; mf < 4; ++mf)
#pragma unroll
        for (int nf = 0; nf < 2; ++nf) {
            const int n = bn + wn + nf * 16 + r16;
            if (n < Nreal) {
                const float bv = bias[n];
#pragma unroll
                for (int r = 0; r < 8; ++r) {
                    const int m = bm + wm + mf * 16 + hi * 8 + r;
                    float v = gamma * acc[mf][nf][r] + bv;
                    if (RELU) v = fmaxf(v, 0.0f);
                    out[(size_t)m * ldc + n] = (OutT)v;
                }
            }
        }
}

// ---------------- host-side orchestration ----------------
extern "C" void kernel_launch(void* const* d_in, const int* in_sizes, int n_in,
                              void* d_out, int out_size, void* d_ws, size_t ws_size,
                              hipStream_t stream) {
    (void)in_sizes; (void)n_in; (void)out_size; (void)ws_size;
    const float* x  = (const float*)d_in[0];
    const float* W1 = (const float*)d_in[1];
    const float* b1 = (const float*)d_in[2];
    const float* W2 = (const float*)d_in[3];
    const float* b2 = (const float*)d_in[4];
    const float* W3 = (const float*)d_in[5];
    const float* b3 = (const float*)d_in[6];
    const float* W4 = (const float*)d_in[7];
    const float* b4 = (const float*)d_in[8];

    const int B = 8192, D = 4096, H = 4096, C = 1000, Cpad = 1024;

    char* ws = (char*)d_ws;
    size_t off = 0;
    auto alloc = [&](size_t bytes) -> void* {
        void* p = ws + off;
        off = (off + bytes + 255) & ~(size_t)255;
        return p;
    };
    float*     gammas   = (float*)alloc(4 * sizeof(float));
    float*     partials = (float*)alloc(4 * 1024 * sizeof(float));
    _Float16*  T1 = (_Float16*)alloc((size_t)H * D * 2);
    _Float16*  T2 = (_Float16*)alloc((size_t)H * H * 2);
    _Float16*  T3 = (_Float16*)alloc((size_t)H * H * 2);
    _Float16*  T4 = (_Float16*)alloc((size_t)Cpad * H * 2);
    _Float16*  Xh = (_Float16*)alloc((size_t)B * D * 2);
    _Float16*  Hh = (_Float16*)alloc((size_t)B * H * 2);

    const long n1 = (long)H * D, n2 = (long)H * H, n3 = (long)H * H, n4 = (long)C * H;

    // gammas (deterministic two-stage reduction)
    absum_partial<<<1024, 256, 0, stream>>>(W1, n1, partials + 0 * 1024);
    absum_partial<<<1024, 256, 0, stream>>>(W2, n2, partials + 1 * 1024);
    absum_partial<<<1024, 256, 0, stream>>>(W3, n3, partials + 2 * 1024);
    absum_partial<<<1024, 256, 0, stream>>>(W4, n4, partials + 3 * 1024);
    finalize_gamma<<<4, 256, 0, stream>>>(partials, gammas,
                                          1.0f / (float)n1, 1.0f / (float)n2,
                                          1.0f / (float)n3, 1.0f / (float)n4);

    // ternary codes in f16 (gamma folded into GEMM epilogue)
    quantize_tern<<<4096, 256, 0, stream>>>(W1, T1, n1, n1, gammas + 0);
    quantize_tern<<<4096, 256, 0, stream>>>(W2, T2, n2, n2, gammas + 1);
    quantize_tern<<<4096, 256, 0, stream>>>(W3, T3, n3, n3, gammas + 2);
    quantize_tern<<<4096, 256, 0, stream>>>(W4, T4, n4, (long)Cpad * H, gammas + 3);

    // activations to f16
    cvt_f16<<<((long)B * D / 4 + 255) / 256, 256, 0, stream>>>(x, Xh, (long)B * D / 4);

    // 4 GEMM layers (ping-pong Xh/Hh)
    dim3 blk(256);
    gemm_tern<true,  _Float16><<<dim3(H    / 128, B / 128), blk, 0, stream>>>(Xh, T1, b1, gammas + 0, Hh, B, H, D, H);
    gemm_tern<true,  _Float16><<<dim3(H    / 128, B / 128), blk, 0, stream>>>(Hh, T2, b2, gammas + 1, Xh, B, H, H, H);
    gemm_tern<true,  _Float16><<<dim3(H    / 128, B / 128), blk, 0, stream>>>(Xh, T3, b3, gammas + 2, Hh, B, H, H, H);
    gemm_tern<false, float   ><<<dim3(Cpad / 128, B / 128), blk, 0, stream>>>(Hh, T4, b4, gammas + 3, (float*)d_out, B, C, H, C);
}
